// Block_558345749133
// MI455X (gfx1250) — matrix operand order repaired and checked by reading, not compiled, not run on hardware
//
#include <hip/hip_runtime.h>
#include <hip/hip_bf16.h>
#include <cstdint>

#define BATCH   8
#define T_NODES 4096
#define C_DIM   384
#define H_HEADS 6
#define F_DIM   64
#define E_EDGES 131072
#define ET      (E_EDGES + T_NODES)   // 135168 (self loops appended)
#define DFF     1536
#define ROWS    (BATCH * T_NODES)     // 32768

typedef __bf16 bf16_t;
typedef __attribute__((ext_vector_type(16))) __bf16 v16bf;
typedef __attribute__((ext_vector_type(8)))  __bf16 v8bf;
typedef __attribute__((ext_vector_type(8)))  float  v8f;
typedef __attribute__((ext_vector_type(4)))  unsigned int v4u;
typedef __attribute__((ext_vector_type(8)))  int v8i;
typedef __attribute__((ext_vector_type(4)))  int v4i;

// Tensor Data Mover path only when the device compiler exposes the builtins.
#if defined(__AMDGCN__) && __has_builtin(__builtin_amdgcn_tensor_load_to_lds) && \
    __has_builtin(__builtin_amdgcn_s_wait_tensorcnt)
#define USE_TDM 1
#else
#define USE_TDM 0
#endif

// ---------------------------------------------------------------------------
// helpers
// ---------------------------------------------------------------------------
__device__ inline float wave_reduce_add(float v) {
#pragma unroll
  for (int off = 16; off > 0; off >>= 1) v += __shfl_xor(v, off, 32);
  return v;
}

__device__ inline void atomicMaxF(float* addr, float v) {
  // classic monotone-bit-pattern trick; buffer initialized to -1e30f
  if (v >= 0.0f) atomicMax((int*)addr, __float_as_int(v));
  else           atomicMin((unsigned int*)addr, __float_as_uint(v));
}

__device__ inline float leaky(float a) { return (a >= 0.0f) ? a : 0.2f * a; }

#if USE_TDM
// Issue a TDM load of a [tile_rows x tile_cols] bf16 tile whose top-left is at
// gbase (row stride = tensor_k elements) into LDS byte offset lds_off.
// D# layout per CDNA5 ISA 8.3/8.4 (2-D tensor, groups 2/3 zero).
__device__ inline void tdm_load_tile_a(const bf16_t* gbase, unsigned lds_off,
                                       unsigned tile_rows, unsigned tile_cols,
                                       unsigned tensor_k, unsigned tensor_m) {
  unsigned long long ga = (unsigned long long)(uintptr_t)gbase;
  v4u g0;
  g0[0] = 1u;                                  // count=1 (valid), user mode
  g0[1] = lds_off;                             // lds_addr (bytes)
  g0[2] = (unsigned)(ga & 0xffffffffu);        // global_addr[31:0]
  g0[3] = (unsigned)((ga >> 32) & 0x01ffffffu) // global_addr[56:32]
          | (2u << 30);                        // type=2 ("image")
  v8i g1;
  g1[0] = (int)(1u << 16);                                   // data_size=1 -> 2B
  g1[1] = (int)((tensor_k & 0xffffu) << 16);                 // tensor_dim0[15:0]
  g1[2] = (int)((tensor_k >> 16) | ((tensor_m & 0xffffu) << 16)); // dim0 hi | dim1 lo
  g1[3] = (int)((tensor_m >> 16) | (tile_cols << 16));       // dim1 hi | tile_dim0
  g1[4] = (int)tile_rows;                                    // tile_dim1 (tile_dim2=0)
  g1[5] = (int)tensor_k;                                     // dim0_stride[31:0]
  g1[6] = 0;                                                 // dim0_stride hi | dim1_stride lo
  g1[7] = 0;
  v4i z4 = {0, 0, 0, 0};
#if defined(__clang_major__) && (__clang_major__ >= 23)
  v8i z8 = {0, 0, 0, 0, 0, 0, 0, 0};
  __builtin_amdgcn_tensor_load_to_lds(g0, g1, z4, z4, z8, 0);
#else
  __builtin_amdgcn_tensor_load_to_lds(g0, g1, z4, z4, 0);
#endif
}
#endif  // USE_TDM

// ---------------------------------------------------------------------------
// fill
// ---------------------------------------------------------------------------
__global__ void fill_f32(float* __restrict__ p, float v, size_t n) {
  for (size_t i = (size_t)blockIdx.x * blockDim.x + threadIdx.x; i < n;
       i += (size_t)gridDim.x * blockDim.x)
    p[i] = v;
}

// W [K][N] (row major, f32) -> Wt [N][K] (row major, bf16)
__global__ void transpose_bf16(const float* __restrict__ W, bf16_t* __restrict__ Wt,
                               int K, int N) {
  size_t idx = (size_t)blockIdx.x * blockDim.x + threadIdx.x;
  if (idx >= (size_t)K * N) return;
  int k = (int)(idx % K);
  int n = (int)(idx / K);
  Wt[idx] = (bf16_t)W[(size_t)k * N + n];
}

// ---------------------------------------------------------------------------
// LayerNorm (one wave32 per row of 384)
// ---------------------------------------------------------------------------
__global__ __launch_bounds__(32)
void ln1_kernel(const float* __restrict__ x, const float* __restrict__ g,
                const float* __restrict__ b, bf16_t* __restrict__ out) {
  const int row  = blockIdx.x;
  const int lane = threadIdx.x;
  float lv[12];
  float s = 0.0f;
#pragma unroll
  for (int i = 0; i < 12; ++i) {
    lv[i] = x[(size_t)row * C_DIM + lane + i * 32];
    s += lv[i];
  }
  s = wave_reduce_add(s);
  const float mu = s * (1.0f / C_DIM);
  float vs = 0.0f;
#pragma unroll
  for (int i = 0; i < 12; ++i) { float d = lv[i] - mu; vs += d * d; }
  vs = wave_reduce_add(vs);
  const float rstd = rsqrtf(vs * (1.0f / C_DIM) + 1e-5f);
#pragma unroll
  for (int i = 0; i < 12; ++i) {
    const int col = lane + i * 32;
    out[(size_t)row * C_DIM + col] = (bf16_t)((lv[i] - mu) * rstd * g[col] + b[col]);
  }
}

// x2 = x + gat + b_gat ; out = LN(x2) in bf16
__global__ __launch_bounds__(32)
void resid_ln2_kernel(const float* __restrict__ x, const float* __restrict__ gat,
                      const float* __restrict__ bgat, const float* __restrict__ g,
                      const float* __restrict__ b, float* __restrict__ x2,
                      bf16_t* __restrict__ out) {
  const int row  = blockIdx.x;
  const int lane = threadIdx.x;
  float lv[12];
  float s = 0.0f;
#pragma unroll
  for (int i = 0; i < 12; ++i) {
    const int col = lane + i * 32;
    const size_t idx = (size_t)row * C_DIM + col;
    float v = x[idx] + gat[idx] + bgat[col];
    x2[idx] = v;
    lv[i] = v;
    s += v;
  }
  s = wave_reduce_add(s);
  const float mu = s * (1.0f / C_DIM);
  float vs = 0.0f;
#pragma unroll
  for (int i = 0; i < 12; ++i) { float d = lv[i] - mu; vs += d * d; }
  vs = wave_reduce_add(vs);
  const float rstd = rsqrtf(vs * (1.0f / C_DIM) + 1e-5f);
#pragma unroll
  for (int i = 0; i < 12; ++i) {
    const int col = lane + i * 32;
    out[(size_t)row * C_DIM + col] = (bf16_t)((lv[i] - mu) * rstd * g[col] + b[col]);
  }
}

// ---------------------------------------------------------------------------
// WMMA bf16 GEMM:  out = act(A @ Bt^T + bias) (+ resid)
//   A  : [M,K] bf16 row major      Bt : [N,K] bf16 row major (i.e. B^T)
//   block = 256 threads = 8 waves; block tile 128(M) x 64(N); K step 32
//   wave (wm 0..3, wn 0..1) owns a 32x32 sub-tile = 2x2 WMMA accumulators
//   A tiles double-buffered in LDS via the Tensor Data Mover (wave 0 issues,
//   TENSORcnt + workgroup barrier publishes); fallback: cooperative loads.
// ---------------------------------------------------------------------------
__global__ __launch_bounds__(256)
void gemm_bf16_wmma(const bf16_t* __restrict__ A, const bf16_t* __restrict__ Bt,
                    const float* __restrict__ bias, const float* __restrict__ resid,
                    float* __restrict__ outF, bf16_t* __restrict__ outB,
                    int M, int N, int K, int relu) {
  __shared__ __align__(16) bf16_t sA[2][128 * 32];   // only LDS alloc -> offset 0

  const int tid  = threadIdx.x;
  const int lane = tid & 31;
  const int wv   = tid >> 5;
  const int wm   = wv >> 1;     // 0..3
  const int wn   = wv & 1;      // 0..1
  const int half = lane >> 4;   // 0/1 (ISA lane-half)
  const int r    = lane & 15;

  const int m0 = blockIdx.y * 128;
  const int n0 = blockIdx.x * 64;

  const v8f vz = {0.f, 0.f, 0.f, 0.f, 0.f, 0.f, 0.f, 0.f};
  v8f acc[2][2];
  acc[0][0] = vz; acc[0][1] = vz; acc[1][0] = vz; acc[1][1] = vz;

#if USE_TDM
  // Never executes (relu is 0/1 at runtime) but the compiler cannot prove it:
  // guarantees sA has a visible potential writer so its loads are not poison.
  if (relu > 1) sA[0][tid] = (bf16_t)1.0f;
  if (wv == 0)
    tdm_load_tile_a(A + (size_t)m0 * K, 0u, 128u, 32u, (unsigned)K, (unsigned)M);
#else
  const int ldr = tid >> 1;        // staging row 0..127
  const int ldc = (tid & 1) * 16;  // staging col 0 / 16
#endif

  for (int k0 = 0; k0 < K; k0 += 32) {
    const int st = (k0 >> 5) & 1;

#if USE_TDM
    // Scalar op: executes regardless of EXEC; only wave 0 has TENSORcnt != 0.
    __builtin_amdgcn_s_wait_tensorcnt(0);
    __syncthreads();  // tile st visible; everyone done reading st^1
    if (wv == 0 && (k0 + 32) < K)
      tdm_load_tile_a(A + (size_t)m0 * K + (k0 + 32),
                      (unsigned)((st ^ 1) * 128 * 32 * 2), 128u, 32u,
                      (unsigned)K, (unsigned)M);
    // The TDM engine writes sA behind the compiler's back. Passing the LDS
    // base INTO the asm makes the address escape, so the memory clobber is a
    // legal may-write to sA: the ds_loads below can be neither undef-folded
    // nor hoisted out of the loop.
    asm volatile("" :: "v"((void*)&sA[0][0]) : "memory");
#else
    const uint4* gsrc = (const uint4*)(A + (size_t)(m0 + ldr) * K + (k0 + ldc));
    uint4* sdst = (uint4*)(&sA[st][ldr * 32 + ldc]);
    sdst[0] = gsrc[0];
    sdst[1] = gsrc[1];
    if (k0 + 32 < K)
      __builtin_prefetch((const void*)(gsrc + 4), 0, 1);
    __syncthreads();
#endif

    // ---- A fragments (ISA 16-bit A 16x32 layout) ----
    v16bf afr[2];
#pragma unroll
    for (int mt = 0; mt < 2; ++mt) {
      const bf16_t* ap = &sA[st][(wm * 32 + mt * 16 + r) * 32 + half * 8];
      v8bf lo = *(const v8bf*)ap;          // K = (half?8:0) + 0..7
      v8bf hi = *(const v8bf*)(ap + 16);   // K = 16 + (half?8:0) + 0..7
      afr[mt] = __builtin_shufflevector(lo, hi, 0, 1, 2, 3, 4, 5, 6, 7,
                                        8, 9, 10, 11, 12, 13, 14, 15);
    }

    // ---- B fragments from global Wt[N][K] (weights are L2-resident) ----
    v16bf bfr[2];
#pragma unroll
    for (int nt = 0; nt < 2; ++nt) {
      const bf16_t* bp =
          Bt + (size_t)(n0 + wn * 32 + nt * 16 + r) * K + k0 + half * 8;
      v8bf lo = *(const v8bf*)bp;        // K = half*8 + 0..7
      v8bf hi = *(const v8bf*)(bp + 16); // K = 16 + half*8 + 0..7
      bfr[nt] = __builtin_shufflevector(lo, hi, 0, 1, 2, 3, 4, 5, 6, 7,
                                        8, 9, 10, 11, 12, 13, 14, 15);
    }

#pragma unroll
    for (int mt = 0; mt < 2; ++mt)
#pragma unroll
      for (int nt = 0; nt < 2; ++nt)
        acc[mt][nt] = __builtin_amdgcn_wmma_f32_16x16x32_bf16(
            false, afr[mt], false, bfr[nt], (short)0, acc[mt][nt], false, false);

#if !USE_TDM
    __syncthreads();
#endif
  }

  // ---- epilogue: C/D layout VGPR i -> row (half*8 + i), col r ----
#pragma unroll
  for (int mt = 0; mt < 2; ++mt) {
#pragma unroll
    for (int nt = 0; nt < 2; ++nt) {
      const int col = n0 + wn * 32 + nt * 16 + r;
      const float bv = bias ? bias[col] : 0.0f;
#pragma unroll
      for (int i = 0; i < 8; ++i) {
        const int row = m0 + wm * 32 + mt * 16 + half * 8 + i;
        float v = acc[mt][nt][i] + bv;
        if (relu) v = fmaxf(v, 0.0f);
        const size_t idx = (size_t)row * N + col;
        if (resid) v += resid[idx];
        if (outF) outF[idx] = v;
        else      outB[idx] = (bf16_t)v;
      }
    }
  }
}

// ---------------------------------------------------------------------------
// attention logits: a_src/a_dst [ROWS,H]
// ---------------------------------------------------------------------------
__global__ void attn_scores(const float* __restrict__ hmat,
                            const float* __restrict__ att_src,
                            const float* __restrict__ att_dst,
                            float* __restrict__ a_src, float* __restrict__ a_dst) {
  const int idx = blockIdx.x * blockDim.x + threadIdx.x;
  if (idx >= ROWS * H_HEADS) return;
  const int row = idx / H_HEADS;
  const int hd  = idx % H_HEADS;
  const float* hp = hmat + (size_t)row * C_DIM + hd * F_DIM;
  const float* as = att_src + hd * F_DIM;
  const float* ad = att_dst + hd * F_DIM;
  float s1 = 0.0f, s2 = 0.0f;
#pragma unroll
  for (int f = 0; f < F_DIM; ++f) {
    const float v = hp[f];
    s1 += v * as[f];
    s2 += v * ad[f];
  }
  a_src[idx] = s1;
  a_dst[idx] = s2;
}

// ---------------------------------------------------------------------------
// edge passes (edges 0..E-1 from edge_index, E..E+T-1 are self-loops)
// ---------------------------------------------------------------------------
__device__ inline void edge_decode(size_t idx, const int* __restrict__ ei,
                                   int& b, int& hd, int& s, int& d) {
  hd = (int)(idx % H_HEADS);
  size_t t = idx / H_HEADS;
  int e = (int)(t % ET);
  b = (int)(t / ET);
  s = (e < E_EDGES) ? ei[e] : (e - E_EDGES);
  d = (e < E_EDGES) ? ei[E_EDGES + e] : (e - E_EDGES);
}

__global__ void edge_max_kernel(const int* __restrict__ ei,
                                const float* __restrict__ a_src,
                                const float* __restrict__ a_dst,
                                float* __restrict__ mmax) {
  const size_t idx = (size_t)blockIdx.x * blockDim.x + threadIdx.x;
  if (idx >= (size_t)BATCH * ET * H_HEADS) return;
  int b, hd, s, d;
  edge_decode(idx, ei, b, hd, s, d);
  const size_t si = ((size_t)(b * T_NODES + s)) * H_HEADS + hd;
  const size_t di = ((size_t)(b * T_NODES + d)) * H_HEADS + hd;
  atomicMaxF(&mmax[di], leaky(a_src[si] + a_dst[di]));
}

__global__ void edge_sum_kernel(const int* __restrict__ ei,
                                const float* __restrict__ a_src,
                                const float* __restrict__ a_dst,
                                const float* __restrict__ mmax,
                                float* __restrict__ den) {
  const size_t idx = (size_t)blockIdx.x * blockDim.x + threadIdx.x;
  if (idx >= (size_t)BATCH * ET * H_HEADS) return;
  int b, hd, s, d;
  edge_decode(idx, ei, b, hd, s, d);
  const size_t si = ((size_t)(b * T_NODES + s)) * H_HEADS + hd;
  const size_t di = ((size_t)(b * T_NODES + d)) * H_HEADS + hd;
  atomicAdd(&den[di], __expf(leaky(a_src[si] + a_dst[di]) - mmax[di]));
}

__global__ void edge_msg_kernel(const int* __restrict__ ei,
                                const float* __restrict__ a_src,
                                const float* __restrict__ a_dst,
                                const float* __restrict__ mmax,
                                const float* __restrict__ den,
                                const float* __restrict__ hmat,
                                float* __restrict__ gat) {
  const size_t idx = (size_t)blockIdx.x * blockDim.x + threadIdx.x;
  if (idx >= (size_t)BATCH * ET * H_HEADS) return;
  int b, hd, s, d;
  edge_decode(idx, ei, b, hd, s, d);
  const size_t si = ((size_t)(b * T_NODES + s)) * H_HEADS + hd;
  const size_t di = ((size_t)(b * T_NODES + d)) * H_HEADS + hd;
  const float w = __expf(leaky(a_src[si] + a_dst[di]) - mmax[di]) / den[di];
  const float4* hs =
      (const float4*)(hmat + ((size_t)(b * T_NODES + s)) * C_DIM + hd * F_DIM);
  float* gp = gat + ((size_t)(b * T_NODES + d)) * C_DIM + hd * F_DIM;
#pragma unroll
  for (int f4 = 0; f4 < F_DIM / 4; ++f4) {
    const float4 hv = hs[f4];
    atomicAdd(&gp[f4 * 4 + 0], w * hv.x);
    atomicAdd(&gp[f4 * 4 + 1], w * hv.y);
    atomicAdd(&gp[f4 * 4 + 2], w * hv.z);
    atomicAdd(&gp[f4 * 4 + 3], w * hv.w);
  }
}

// ---------------------------------------------------------------------------
// launch
// ---------------------------------------------------------------------------
extern "C" void kernel_launch(void* const* d_in, const int* in_sizes, int n_in,
                              void* d_out, int out_size, void* d_ws, size_t ws_size,
                              hipStream_t stream) {
  (void)in_sizes; (void)n_in; (void)out_size; (void)ws_size;
  const float* x    = (const float*)d_in[0];
  const int*   ei   = (const int*)d_in[1];
  const float* Wg   = (const float*)d_in[2];
  const float* asrc = (const float*)d_in[3];
  const float* adst = (const float*)d_in[4];
  const float* bgat = (const float*)d_in[5];
  const float* g1   = (const float*)d_in[6];
  const float* b1n  = (const float*)d_in[7];
  const float* g2   = (const float*)d_in[8];
  const float* b2n  = (const float*)d_in[9];
  const float* W1   = (const float*)d_in[10];
  const float* b1f  = (const float*)d_in[11];
  const float* W2   = (const float*)d_in[12];
  const float* b2f  = (const float*)d_in[13];
  float* out = (float*)d_out;

  uint8_t* p = (uint8_t*)d_ws;
  auto take = [&](size_t bytes) -> void* {
    void* r = (void*)p;
    p += (bytes + 255) & ~(size_t)255;
    return r;
  };

  bf16_t* lnbf = (bf16_t*)take((size_t)ROWS * C_DIM * sizeof(bf16_t));  // ln1 then ln2
  bf16_t* Wgt  = (bf16_t*)take((size_t)C_DIM * C_DIM * sizeof(bf16_t));
  bf16_t* W1t  = (bf16_t*)take((size_t)C_DIM * DFF * sizeof(bf16_t));
  bf16_t* W2t  = (bf16_t*)take((size_t)DFF * C_DIM * sizeof(bf16_t));
  float*  hbuf = (float*)take((size_t)ROWS * C_DIM * sizeof(float));
  float*  a_s  = (float*)take((size_t)ROWS * H_HEADS * sizeof(float));
  float*  a_d  = (float*)take((size_t)ROWS * H_HEADS * sizeof(float));
  float*  mmax = (float*)take((size_t)ROWS * H_HEADS * sizeof(float));
  float*  den  = (float*)take((size_t)ROWS * H_HEADS * sizeof(float));
  float*  gat  = (float*)take((size_t)ROWS * C_DIM * sizeof(float));
  float*  x2   = (float*)take((size_t)ROWS * C_DIM * sizeof(float));
  bf16_t* ffn1 = (bf16_t*)take((size_t)ROWS * DFF * sizeof(bf16_t));

  // weights -> bf16, transposed to [N][K]
  transpose_bf16<<<(C_DIM * C_DIM + 255) / 256, 256, 0, stream>>>(Wg, Wgt, C_DIM, C_DIM);
  transpose_bf16<<<(C_DIM * DFF + 255) / 256, 256, 0, stream>>>(W1, W1t, C_DIM, DFF);
  transpose_bf16<<<(DFF * C_DIM + 255) / 256, 256, 0, stream>>>(W2, W2t, DFF, C_DIM);

  // LN1 -> bf16
  ln1_kernel<<<ROWS, 32, 0, stream>>>(x, g1, b1n, lnbf);

  // h = ln1 @ W_gat   (M=32768, N=384, K=384)
  gemm_bf16_wmma<<<dim3(C_DIM / 64, ROWS / 128), 256, 0, stream>>>(
      lnbf, Wgt, nullptr, nullptr, hbuf, nullptr, ROWS, C_DIM, C_DIM, 0);

  // attention logits
  attn_scores<<<(ROWS * H_HEADS + 255) / 256, 256, 0, stream>>>(hbuf, asrc, adst, a_s, a_d);

  // init segment buffers
  fill_f32<<<512, 256, 0, stream>>>(mmax, -1e30f, (size_t)ROWS * H_HEADS);
  fill_f32<<<512, 256, 0, stream>>>(den, 0.0f, (size_t)ROWS * H_HEADS);
  fill_f32<<<4096, 256, 0, stream>>>(gat, 0.0f, (size_t)ROWS * C_DIM);

  // edge softmax + aggregation
  const size_t etot = (size_t)BATCH * ET * H_HEADS;
  const int eblk = (int)((etot + 255) / 256);
  edge_max_kernel<<<eblk, 256, 0, stream>>>(ei, a_s, a_d, mmax);
  edge_sum_kernel<<<eblk, 256, 0, stream>>>(ei, a_s, a_d, mmax, den);
  edge_msg_kernel<<<eblk, 256, 0, stream>>>(ei, a_s, a_d, mmax, den, hbuf, gat);

  // x2 = x + gat + b_gat ; lnbf = LN2(x2)
  resid_ln2_kernel<<<ROWS, 32, 0, stream>>>(x, gat, bgat, g2, b2n, x2, lnbf);

  // ffn1 = relu(ln2 @ W1 + b1)  (N=1536, K=384), bf16 out
  gemm_bf16_wmma<<<dim3(DFF / 64, ROWS / 128), 256, 0, stream>>>(
      lnbf, W1t, b1f, nullptr, nullptr, ffn1, ROWS, DFF, C_DIM, 1);

  // out = x2 + ffn1 @ W2 + b2   (N=384, K=1536), f32 out
  gemm_bf16_wmma<<<dim3(C_DIM / 64, ROWS / 128), 256, 0, stream>>>(
      ffn1, W2t, b2f, x2, out, nullptr, ROWS, C_DIM, DFF, 0);
}
